// SNNWithSTDP_69140383531094
// MI455X (gfx1250) — matrix operand order repaired and checked by reading, not compiled, not run on hardware
//
#include <hip/hip_runtime.h>
#include <stdint.h>

// ---- model config (mirrors reference) ----
#define BATCH 128
#define TSTEPS 50
#define D_IN 1024
#define H1 2048
#define H2 2048
#define D_OUT 128
#define ALPHA 0.9f
#define V_TH 1.0f

typedef __attribute__((ext_vector_type(16))) _Float16 v16h;
typedef __attribute__((ext_vector_type(8)))  float    v8f;

union HalfVec { v16h v; uint4 u[2]; };

// ---------------- helper kernels ----------------

__global__ __launch_bounds__(256) void zero_f32(float* __restrict__ p, int n) {
    int i = blockIdx.x * 256 + threadIdx.x;
    if (i < n) p[i] = 0.0f;
}

__global__ __launch_bounds__(256) void f32_to_f16(const float* __restrict__ src,
                                                  _Float16* __restrict__ dst, int n) {
    int i = blockIdx.x * 256 + threadIdx.x;
    if (i < n) dst[i] = (_Float16)src[i];
}

// Poisson/rate encoding: spikes[t][b][d] = (noise[b][t][d] < x[b][d]) ? 1 : 0  (exact in f16)
__global__ __launch_bounds__(256) void encode_spikes(const float* __restrict__ noise,
                                                     const float* __restrict__ x,
                                                     _Float16* __restrict__ spikes) {
    int i = blockIdx.x * 256 + threadIdx.x;
    const int total = BATCH * TSTEPS * D_IN;
    if (i >= total) return;
    int d = i % D_IN;
    int t = (i / D_IN) % TSTEPS;
    int b = i / (D_IN * TSTEPS);
    float sp = (noise[i] < x[b * D_IN + d]) ? 1.0f : 0.0f;
    spikes[((size_t)t * BATCH + b) * D_IN + d] = (_Float16)sp;
}

// ---------------- fused GEMM + LIF step (barrier-free, register-resident B) ----
// out[b][n] = sum_k s_in[b][k] * W[n][k]   (W in torch Linear layout [N,K], f16)
// v = ALPHA*v + out + bias ; s = (v >= V_TH) ; v -= s ; optional spike-count accum.
//
// One wave32 owns one 16-neuron N-column and MT 16-row M-tiles (batch=128 -> 8
// M-tiles total, WPN = 8/MT waves share a column). Per k-step: issue ALL
// fragment loads (B once + MT A-frags) as one clause group, then MT independent
// WMMAs — only the first WMMA of the group pays the L2 load wait; the compiler's
// unroll can hoist the next group's loads above the current WMMA block.
template<int K, int N, int MT, bool ACC>
__global__ __launch_bounds__(256) void lif_gemm(const _Float16* __restrict__ s_in,
                                                const _Float16* __restrict__ Wh,
                                                const float* __restrict__ bias,
                                                float* __restrict__ v,
                                                _Float16* __restrict__ s_out,
                                                float* __restrict__ out_acc) {
    constexpr int WPN = 8 / MT;                 // waves sharing one N column
    const int tid  = threadIdx.x;
    const int wave = tid >> 5;
    const int lane = tid & 31;
    const int gw   = blockIdx.x * 8 + wave;     // global wave id
    const int ntile = gw / WPN;
    const int mt0   = (gw % WPN) * MT;
    const int nbase = ntile * 16;

    // Fragment addressing per ISA 7.12.2 (16-bit, wave32):
    // A: lanes 0-15 hold K{0..7,16..23}, lanes 16-31 hold K{8..15,24..31}; M = lane&15
    // B: lanes 0-15 hold K{0..7,16..23}, lanes 16-31 hold K{8..15,24..31}; N = lane&15
    const int m     = lane & 15;
    const int khalf = lane >> 4;                // 0 or 1
    const _Float16* abase = s_in + (size_t)(mt0 * 16 + m) * K + khalf * 8;
    const _Float16* brow  = Wh + (size_t)(nbase + (lane & 15)) * K + khalf * 8;

    v8f c[MT];
    #pragma unroll
    for (int i = 0; i < MT; ++i) c[i] = (v8f){};

    #pragma unroll 2
    for (int k0 = 0; k0 < K; k0 += 32) {
        // ---- load phase: one clause group, no WMMA interleaved ----
        HalfVec b, a[MT];
        b.u[0] = *(const uint4*)(brow + k0);        // K chunk {0..7} / {8..15}
        b.u[1] = *(const uint4*)(brow + k0 + 16);   // K chunk {16..23} / {24..31}
        #pragma unroll
        for (int i = 0; i < MT; ++i) {
            const _Float16* ap = abase + (size_t)i * 16 * K + k0;
            a[i].u[0] = *(const uint4*)(ap);        // K chunk {0..7} / {8..15}
            a[i].u[1] = *(const uint4*)(ap + 16);   // K chunk {16..23} / {24..31}
        }
        // ---- compute phase: MT independent accumulators ----
        #pragma unroll
        for (int i = 0; i < MT; ++i) {
            c[i] = __builtin_amdgcn_wmma_f32_16x16x32_f16(
                       /*neg_a=*/false, a[i].v, /*neg_b=*/false, b.v,
                       /*c_mod=*/(short)0, c[i], /*reuse_a=*/false, /*reuse_b=*/false);
        }
    }

    // Fused LIF epilogue. C layout: VGPR r -> M=r (lanes 0-15) / M=r+8 (lanes 16-31); N=lane&15
    const int col  = nbase + (lane & 15);
    const float bcol = bias[col];
    #pragma unroll
    for (int i = 0; i < MT; ++i) {
        const int rbase = (mt0 + i) * 16 + 8 * khalf;
        #pragma unroll
        for (int r = 0; r < 8; ++r) {
            const int row = rbase + r;
            const size_t idx = (size_t)row * N + col;
            float vv = ALPHA * v[idx] + c[i][r] + bcol;
            float s  = (vv >= V_TH) ? 1.0f : 0.0f;
            v[idx] = vv - s;                        // soft reset
            s_out[idx] = (_Float16)s;
            if constexpr (ACC) out_acc[idx] += s;   // N == D_OUT on the ACC layer
        }
    }
}

// ---------------- host-side launcher ----------------

extern "C" void kernel_launch(void* const* d_in, const int* in_sizes, int n_in,
                              void* d_out, int out_size, void* d_ws, size_t ws_size,
                              hipStream_t stream) {
    const float* x     = (const float*)d_in[0];
    const float* noise = (const float*)d_in[1];
    const float* W1    = (const float*)d_in[2];
    const float* b1    = (const float*)d_in[3];
    const float* W2    = (const float*)d_in[4];
    const float* b2    = (const float*)d_in[5];
    const float* W3    = (const float*)d_in[6];
    const float* b3    = (const float*)d_in[7];
    float* out = (float*)d_out;

    // workspace carve-out (~29 MB), 256B aligned slices
    char* ws = (char*)d_ws;
    size_t off = 0;
    auto carve = [&](size_t bytes) -> char* {
        char* p = ws + off;
        off += (bytes + 255) & ~(size_t)255;
        return p;
    };
    _Float16* spikes = (_Float16*)carve((size_t)TSTEPS * BATCH * D_IN * 2);
    _Float16* W1h    = (_Float16*)carve((size_t)H1 * D_IN * 2);
    _Float16* W2h    = (_Float16*)carve((size_t)H2 * H1 * 2);
    _Float16* W3h    = (_Float16*)carve((size_t)D_OUT * H2 * 2);
    _Float16* s1     = (_Float16*)carve((size_t)BATCH * H1 * 2);
    _Float16* s2     = (_Float16*)carve((size_t)BATCH * H2 * 2);
    _Float16* s3     = (_Float16*)carve((size_t)BATCH * D_OUT * 2);
    float*    v1     = (float*)   carve((size_t)BATCH * H1 * 4);
    float*    v2     = (float*)   carve((size_t)BATCH * H2 * 4);
    float*    v3     = (float*)   carve((size_t)BATCH * D_OUT * 4);

    auto gridFor = [](int n) { return dim3((unsigned)((n + 255) / 256)); };

    // deterministic init each call: membranes and output spike counts start at zero
    zero_f32<<<gridFor(BATCH * H1),    256, 0, stream>>>(v1,  BATCH * H1);
    zero_f32<<<gridFor(BATCH * H2),    256, 0, stream>>>(v2,  BATCH * H2);
    zero_f32<<<gridFor(BATCH * D_OUT), 256, 0, stream>>>(v3,  BATCH * D_OUT);
    zero_f32<<<gridFor(BATCH * D_OUT), 256, 0, stream>>>(out, BATCH * D_OUT);

    // weight down-conversion (L2-resident afterwards; spikes are exact in f16)
    f32_to_f16<<<gridFor(H1 * D_IN),  256, 0, stream>>>(W1, W1h, H1 * D_IN);
    f32_to_f16<<<gridFor(H2 * H1),    256, 0, stream>>>(W2, W2h, H2 * H1);
    f32_to_f16<<<gridFor(D_OUT * H2), 256, 0, stream>>>(W3, W3h, D_OUT * H2);

    // rate encoding for all timesteps
    encode_spikes<<<gridFor(BATCH * TSTEPS * D_IN), 256, 0, stream>>>(noise, x, spikes);

    // sequential scan over timesteps; 3 fused GEMM+LIF layers per step
    // layers 1/2: 128 N-tiles, MT=4 -> 256 waves = 32 blocks (more WGPs in flight)
    // layer  3:     8 N-tiles, MT=1 -> 64 waves  =  8 blocks (spread the tiny GEMM)
    for (int t = 0; t < TSTEPS; ++t) {
        const _Float16* st = spikes + (size_t)t * BATCH * D_IN;
        lif_gemm<D_IN, H1,    4, false><<<32, 256, 0, stream>>>(st, W1h, b1, v1, s1, nullptr);
        lif_gemm<H1,   H2,    4, false><<<32, 256, 0, stream>>>(s1, W2h, b2, v2, s2, nullptr);
        lif_gemm<H2,   D_OUT, 1, true ><<< 8, 256, 0, stream>>>(s2, W3h, b3, v3, s3, out);
    }
}
